// Backbone_71064528879926
// MI455X (gfx1250) — compile-verified, matrix operand order checked by reading, not run on hardware
//
#include <hip/hip_runtime.h>
#include <hip/hip_bf16.h>

// ---------------------------------------------------------------------------
// MI455X (gfx1250, wave32) Point-Transformer backbone.
// Dense math: bf16-WMMA GEMM with LDS-staged A tiles and pre-swizzled bf16
// weights (B operand loaded in register layout, coalesced b128).
// Epilogue variants (bias/residual/relu) compiled as templates: no runtime
// pointer-null branches in device code.
// Irregular ops (KNN / FPS / gathers / softmax / BN / maxpool): wave32 VALU.
// ---------------------------------------------------------------------------

typedef __bf16 bf16_t;
typedef __attribute__((ext_vector_type(16))) __bf16 v16bf;
typedef __attribute__((ext_vector_type(8)))  float  v8f;

static inline unsigned cdivu(size_t a, unsigned b) { return (unsigned)((a + (size_t)b - 1) / (size_t)b); }

// ---------------------------------------------------------------------------
// Weight pre-swizzle: W fp32 [Kd x N] -> bf16 in the exact B-operand register
// layout. Out element index: ((tn*nChunks + ch)*32 + lane)*16 + e, where for
// lane l: col = tn*16 + (l&15), kb = (l>>4)*8, j = e>>1,
//   koff = (j<4 ? kb+2j : 16+kb+2(j-4)) + (e&1),  k = ch*32 + koff.
// Out-of-range K zero-filled so the GEMM needs no guards.
// ---------------------------------------------------------------------------
__global__ void wswizzle_kernel(const float* __restrict__ W, bf16_t* __restrict__ out,
                                int Kd, int N, int nChunks, size_t total)
{
  size_t t = (size_t)blockIdx.x * blockDim.x + threadIdx.x;
  if (t >= total) return;
  int e    = (int)(t & 15);
  int lane = (int)((t >> 4) & 31);
  size_t rc = t >> 9;
  int ch = (int)(rc % nChunks);
  int tn = (int)(rc / nChunks);
  int l16 = lane & 15, kb = (lane >> 4) * 8;
  int j = e >> 1;
  int koff = ((j < 4) ? (kb + 2 * j) : (16 + kb + 2 * (j - 4))) + (e & 1);
  int k = ch * 32 + koff;
  int col = tn * 16 + l16;
  float v = (k < Kd) ? W[(size_t)k * N + col] : 0.f;
  out[t] = (bf16_t)v;
}

// ---------------------------------------------------------------------------
// GEMM: C[M,N] = act( A[M,K] @ W[K,N] [+ bias] [+ resid] )
// grid = (ceil(tilesN/8), tilesM), block = 256 (8 waves). All 8 waves share
// the same 16 A rows, staged through LDS (row stride 34 words: 8B-aligned
// float2 reads, conflict-free across the 64 banks). B loaded pre-swizzled
// bf16, 32 contiguous bytes per lane per chunk. One v_wmma per K-chunk.
// Epilogue behavior fixed at compile time (HASB/HASR/RELU).
// ---------------------------------------------------------------------------
template <bool HASB, bool HASR, bool RELU>
__global__ __launch_bounds__(256) void gemm_bf16_wmma_kernel(
    const float* __restrict__ A, const bf16_t* __restrict__ Wsw,
    const float* __restrict__ bias, const float* __restrict__ resid,
    float* __restrict__ C, int M, int Kd, int N, int nChunks)
{
  __shared__ float As[16][34];

  const int tid  = threadIdx.x;
  const int lane = tid & 31;
  const int wave = tid >> 5;
  const int tilesN = N >> 4;
  const int tm = blockIdx.y;
  const int tn = blockIdx.x * 8 + wave;
  const bool valid = (tn < tilesN);
  const int l16 = lane & 15;
  const int hi  = lane >> 4;
  const int kb  = hi * 8;

  // cooperative A-stage mapping: 256 threads x 2 floats = 16x32 tile
  const int arow = tid >> 4;          // 0..15
  const int acol = (tid & 15) * 2;    // 0,2,..,30
  const float* Abase = A + (size_t)(tm * 16 + arow) * Kd;
  const bf16_t* Bbase = valid ? (Wsw + (size_t)tn * nChunks * 512 + (size_t)lane * 16)
                              : Wsw;

  v8f acc = {0.f, 0.f, 0.f, 0.f, 0.f, 0.f, 0.f, 0.f};

  for (int ch = 0; ch < nChunks; ++ch) {
    const int k0 = ch * 32;
    float f0, f1;
    if (k0 + 32 <= Kd) {                          // uniform fast path: no guards
      float2 v = *(const float2*)&Abase[k0 + acol];
      f0 = v.x; f1 = v.y;
      if (k0 + 64 <= Kd) __builtin_prefetch(&Abase[k0 + 32 + acol], 0, 1);
    } else {                                      // uniform K-tail: zero fill
      f0 = (k0 + acol     < Kd) ? Abase[k0 + acol]     : 0.f;
      f1 = (k0 + acol + 1 < Kd) ? Abase[k0 + acol + 1] : 0.f;
    }
    __syncthreads();                              // previous chunk consumed
    As[arow][acol] = f0;
    As[arow][acol + 1] = f1;
    __syncthreads();                              // tile visible

    if (valid) {
      v16bf b = *(const v16bf*)(Bbase + (size_t)ch * 512);
      v16bf a;
#pragma unroll
      for (int j = 0; j < 8; ++j) {
        int koff = (j < 4) ? (kb + 2 * j) : (16 + kb + 2 * (j - 4));
        float2 f = *(const float2*)&As[l16][koff];
        a[2 * j] = (bf16_t)f.x; a[2 * j + 1] = (bf16_t)f.y;
      }
      acc = __builtin_amdgcn_wmma_f32_16x16x32_bf16(
          /*neg_a=*/false, a, /*neg_b=*/false, b,
          /*c_mod=*/(short)0, acc, /*reuse_a=*/false, /*reuse_b=*/false);
    }
  }

  if (!valid) return;
  const int col = tn * 16 + l16;
  const float bv = HASB ? bias[col] : 0.f;
  const size_t base = (size_t)(tm * 16 + 8 * hi) * N + col;

  float rv[8];
  if (HASR) {
#pragma unroll
    for (int r = 0; r < 8; ++r) rv[r] = resid[base + (size_t)r * N];
  }
#pragma unroll
  for (int r = 0; r < 8; ++r) {
    float v = acc[r] + bv;
    if (HASR) v += rv[r];
    if (RELU) v = fmaxf(v, 0.f);
    C[base + (size_t)r * N] = v;
  }
}

// --------------------------- KNN (2-D points, K=16) -------------------------
__global__ void knn_kernel(const float* __restrict__ qxy, const float* __restrict__ pxy,
                           int* __restrict__ idx, int M, int N, int BM)
{
  int t = blockIdx.x * blockDim.x + threadIdx.x;
  if (t >= BM) return;
  int b = t / M;
  float qx = qxy[(size_t)t * 2 + 0], qy = qxy[(size_t)t * 2 + 1];
  float bd[16]; int bi[16];
#pragma unroll
  for (int i = 0; i < 16; ++i) { bd[i] = 3.0e38f; bi[i] = 0; }
  const float* p = pxy + (size_t)b * N * 2;
  for (int n = 0; n < N; ++n) {
    float dx = qx - p[2 * n], dy = qy - p[2 * n + 1];
    float d = dx * dx + dy * dy;
    if (d < bd[15]) {
      int j = 15;
      while (j > 0 && bd[j - 1] > d) { bd[j] = bd[j - 1]; bi[j] = bi[j - 1]; --j; }
      bd[j] = d; bi[j] = n;
    }
  }
#pragma unroll
  for (int i = 0; i < 16; ++i) idx[(size_t)t * 16 + i] = bi[i];
}

// ------------------------ Farthest point sampling ---------------------------
__global__ __launch_bounds__(256) void fps_kernel(const float* __restrict__ xy,
                                                  int* __restrict__ out,
                                                  float* __restrict__ dist,
                                                  int N, int npoint)
{
  int b = blockIdx.x;
  const float* p = xy + (size_t)b * N * 2;
  float* d = dist + (size_t)b * N;
  for (int n = threadIdx.x; n < N; n += blockDim.x) d[n] = 1e10f;
  __shared__ int s_far;
  __shared__ float s_rd[256];
  __shared__ int   s_ri[256];
  if (threadIdx.x == 0) s_far = 0;
  __syncthreads();
  for (int it = 0; it < npoint; ++it) {
    int far = s_far;
    if (threadIdx.x == 0) out[(size_t)b * npoint + it] = far;
    float cx = p[2 * far], cy = p[2 * far + 1];
    float bm = -1.f; int bidx = 0;
    for (int n = threadIdx.x; n < N; n += blockDim.x) {
      float dx = p[2 * n] - cx, dy = p[2 * n + 1] - cy;
      float nd = fminf(d[n], dx * dx + dy * dy);
      d[n] = nd;
      if (nd > bm) { bm = nd; bidx = n; }
    }
    s_rd[threadIdx.x] = bm; s_ri[threadIdx.x] = bidx;
    __syncthreads();
    for (int s = blockDim.x / 2; s > 0; s >>= 1) {
      if (threadIdx.x < s && s_rd[threadIdx.x + s] > s_rd[threadIdx.x]) {
        s_rd[threadIdx.x] = s_rd[threadIdx.x + s];
        s_ri[threadIdx.x] = s_ri[threadIdx.x + s];
      }
      __syncthreads();
    }
    if (threadIdx.x == 0) s_far = s_ri[0];
    __syncthreads();
  }
}

// ------------------------------ Gathers -------------------------------------
__global__ void gather_rows_kernel(const float* __restrict__ src, const int* __restrict__ idx,
                                   float* __restrict__ out, int M, int C, int Nsrc, size_t total)
{
  size_t t = (size_t)blockIdx.x * blockDim.x + threadIdx.x;
  if (t >= total) return;
  int c = (int)(t % C);
  size_t r = t / C;                 // over B*M
  int b = (int)(r / M);
  int id = idx[r];
  out[t] = src[((size_t)b * Nsrc + id) * C + c];
}

__global__ void gather_feats_kernel(const float* __restrict__ feats, const int* __restrict__ idx,
                                    float* __restrict__ out, int M, int N, int C, int Cout,
                                    int coff, size_t total)
{
  size_t t = (size_t)blockIdx.x * blockDim.x + threadIdx.x;
  if (t >= total) return;
  int c = (int)(t % C);
  size_t r = t / C;                 // over B*M*K
  int k = (int)(r & 15);
  size_t bm = r >> 4;
  int b = (int)(bm / M);
  int id = idx[bm * 16 + k];
  out[r * Cout + coff + c] = feats[((size_t)b * N + id) * C + c];
}

// out[b,m,k,c] = sign * ( xy[idx] - refxy[b,m] ), c in {0,1}, row stride Cout
__global__ void gather_relxy_kernel(const float* __restrict__ xy, const int* __restrict__ idx,
                                    const float* __restrict__ refxy, float* __restrict__ out,
                                    int M, int N, int Cout, float sign, size_t total)
{
  size_t t = (size_t)blockIdx.x * blockDim.x + threadIdx.x;
  if (t >= total) return;
  int c = (int)(t & 1);
  size_t r = t >> 1;                // over B*M*K
  int k = (int)(r & 15);
  size_t bm = r >> 4;
  int b = (int)(bm / M);
  int id = idx[bm * 16 + k];
  out[r * Cout + c] = sign * (xy[((size_t)b * N + id) * 2 + c] - refxy[bm * 2 + c]);
}

// ----------------------- elementwise / attention ----------------------------
// t[bn,k,c] = q[bn,c] - t[bn,k,c] + pos[bn,k,c]   (in place on t)
__global__ void combine_qkp_kernel(const float* __restrict__ q, float* __restrict__ tt,
                                   const float* __restrict__ pos, int C, size_t total)
{
  size_t t = (size_t)blockIdx.x * blockDim.x + threadIdx.x;
  if (t >= total) return;
  int c = (int)(t % C);
  size_t bn = t / ((size_t)C * 16);
  tt[t] = q[bn * C + c] - tt[t] + pos[t];
}

// res[bn,c] = sum_k softmax_k(attn[bn,k,c]/16) * (v[bn,k,c] + pos[bn,k,c])
__global__ void softmax_wsum_kernel(const float* __restrict__ attn, const float* __restrict__ v,
                                    const float* __restrict__ pos, float* __restrict__ res,
                                    int C, float invsq, size_t total)
{
  size_t t = (size_t)blockIdx.x * blockDim.x + threadIdx.x;
  if (t >= total) return;
  int c = (int)(t % C);
  size_t bn = t / C;
  size_t base = bn * 16 * (size_t)C + c;
  float mx = -3.0e38f;
#pragma unroll
  for (int k = 0; k < 16; ++k) mx = fmaxf(mx, attn[base + (size_t)k * C] * invsq);
  float se = 0.f, acc = 0.f;
#pragma unroll
  for (int k = 0; k < 16; ++k) {
    size_t o = base + (size_t)k * C;
    float e = __expf(attn[o] * invsq - mx);
    se  += e;
    acc += e * (v[o] + pos[o]);
  }
  res[bn * C + c] = acc / se;
}

// ------------------------------ BatchNorm -----------------------------------
__global__ __launch_bounds__(256) void bn_stats_kernel(const float* __restrict__ h, int R, int C,
                                                       float* __restrict__ mean, float* __restrict__ rstd)
{
  int c = blockIdx.x;
  float s = 0.f, s2 = 0.f;
  for (int r = threadIdx.x; r < R; r += blockDim.x) {
    float v = h[(size_t)r * C + c];
    s += v; s2 += v * v;
  }
  __shared__ float ss[256], ss2[256];
  ss[threadIdx.x] = s; ss2[threadIdx.x] = s2;
  __syncthreads();
  for (int st = blockDim.x / 2; st > 0; st >>= 1) {
    if (threadIdx.x < st) { ss[threadIdx.x] += ss[threadIdx.x + st]; ss2[threadIdx.x] += ss2[threadIdx.x + st]; }
    __syncthreads();
  }
  if (threadIdx.x == 0) {
    float mu = ss[0] / R;
    float var = ss2[0] / R - mu * mu;
    mean[c] = mu;
    rstd[c] = rsqrtf(var + 1e-5f);
  }
}

__global__ void bn_apply_relu_kernel(float* __restrict__ h, const float* __restrict__ mean,
                                     const float* __restrict__ rstd, const float* __restrict__ gamma,
                                     const float* __restrict__ beta, int C, size_t total)
{
  size_t t = (size_t)blockIdx.x * blockDim.x + threadIdx.x;
  if (t >= total) return;
  int c = (int)(t % C);
  float v = (h[t] - mean[c]) * rstd[c] * gamma[c] + beta[c];
  h[t] = v > 0.f ? v : 0.f;
}

// ----------------------------- misc small -----------------------------------
__global__ void maxk_kernel(const float* __restrict__ f, float* __restrict__ out, int C, size_t total)
{
  size_t t = (size_t)blockIdx.x * blockDim.x + threadIdx.x;
  if (t >= total) return;
  int c = (int)(t % C);
  size_t bm = t / C;
  size_t base = bm * 16 * (size_t)C + c;
  float mx = -3.0e38f;
#pragma unroll
  for (int k = 0; k < 16; ++k) mx = fmaxf(mx, f[base + (size_t)k * C]);
  out[t] = mx;
}

__global__ void extract_xy_kernel(const float* __restrict__ x, float* __restrict__ xy,
                                  int F, size_t total)
{
  size_t t = (size_t)blockIdx.x * blockDim.x + threadIdx.x;
  if (t >= total) return;
  xy[t] = x[(t >> 1) * F + (t & 1)];
}

// ===========================================================================
// Host orchestration
// ===========================================================================
struct LinP  { const float *W, *b; };
struct ConvP { const float *W, *b, *beta, *gamma; };
struct TFP   { LinP delta1, delta2, fc1, fc2, gamma1, gamma2; const float *wk, *wq, *wv; };

struct Scratch {
  float *X, *Q, *KF, *VF, *Rel;
  float *P0, *P1, *P2, *P3;
  int   *idx;
  float *mean, *rstd;
  bf16_t *Wsw;
};

static void launch_gemm(const float* A, const float* W, const float* bias, const float* resid,
                        float* C, int M, int Kd, int N, int relu, bf16_t* Wsw, hipStream_t s)
{
  int tilesN = N / 16;
  int tilesM = M / 16;
  int nChunks = (Kd + 31) / 32;
  size_t swTotal = (size_t)tilesN * nChunks * 512;
  wswizzle_kernel<<<cdivu(swTotal, 256), 256, 0, s>>>(W, Wsw, Kd, N, nChunks, swTotal);
  dim3 grid(cdivu((size_t)tilesN, 8), (unsigned)tilesM);
  if (bias) {
    if (resid)      gemm_bf16_wmma_kernel<true,  true,  false><<<grid, 256, 0, s>>>(A, Wsw, bias, resid, C, M, Kd, N, nChunks);
    else if (relu)  gemm_bf16_wmma_kernel<true,  false, true ><<<grid, 256, 0, s>>>(A, Wsw, bias, resid, C, M, Kd, N, nChunks);
    else            gemm_bf16_wmma_kernel<true,  false, false><<<grid, 256, 0, s>>>(A, Wsw, bias, resid, C, M, Kd, N, nChunks);
  } else {
    gemm_bf16_wmma_kernel<false, false, false><<<grid, 256, 0, s>>>(A, Wsw, bias, resid, C, M, Kd, N, nChunks);
  }
}

static void run_transformer(const TFP& p, const float* xy, const float* fin, float* fout,
                            int Nn, int dp, Scratch& S, hipStream_t s)
{
  const int Bc = 4, Dd = 256;
  int BN = Bc * Nn;
  knn_kernel<<<cdivu((size_t)BN, 128), 128, 0, s>>>(xy, xy, S.idx, Nn, Nn, BN);

  launch_gemm(fin, p.fc1.W, p.fc1.b, nullptr, S.X, BN, dp, Dd, 0, S.Wsw, s);
  launch_gemm(S.X, p.wq, nullptr, nullptr, S.Q,  BN, Dd, Dd, 0, S.Wsw, s);
  launch_gemm(S.X, p.wk, nullptr, nullptr, S.KF, BN, Dd, Dd, 0, S.Wsw, s);
  launch_gemm(S.X, p.wv, nullptr, nullptr, S.VF, BN, Dd, Dd, 0, S.Wsw, s);

  size_t tot = (size_t)BN * 16 * Dd;
  gather_feats_kernel<<<cdivu(tot, 256), 256, 0, s>>>(S.KF, S.idx, S.P2, Nn, Nn, Dd, Dd, 0, tot);
  gather_feats_kernel<<<cdivu(tot, 256), 256, 0, s>>>(S.VF, S.idx, S.P3, Nn, Nn, Dd, Dd, 0, tot);

  size_t trel = (size_t)BN * 16 * 2;
  gather_relxy_kernel<<<cdivu(trel, 256), 256, 0, s>>>(xy, S.idx, xy, S.Rel, Nn, Nn, 2, -1.f, trel);

  launch_gemm(S.Rel, p.delta1.W, p.delta1.b, nullptr, S.P1, BN * 16, 2,  Dd, 1, S.Wsw, s); // relu
  launch_gemm(S.P1,  p.delta2.W, p.delta2.b, nullptr, S.P0, BN * 16, Dd, Dd, 0, S.Wsw, s); // pos

  combine_qkp_kernel<<<cdivu(tot, 256), 256, 0, s>>>(S.Q, S.P2, S.P0, Dd, tot);              // t in P2
  launch_gemm(S.P2, p.gamma1.W, p.gamma1.b, nullptr, S.P1, BN * 16, Dd, Dd, 1, S.Wsw, s);    // relu
  launch_gemm(S.P1, p.gamma2.W, p.gamma2.b, nullptr, S.P2, BN * 16, Dd, Dd, 0, S.Wsw, s);    // attn

  size_t tres = (size_t)BN * Dd;
  softmax_wsum_kernel<<<cdivu(tres, 256), 256, 0, s>>>(S.P2, S.P3, S.P0, S.X, Dd, 1.0f / 16.0f, tres);

  launch_gemm(S.X, p.fc2.W, p.fc2.b, fin, fout, BN, Dd, dp, 0, S.Wsw, s);  // + residual feats
}

static void run_td(const ConvP* cv, const float* xy, const float* pts, int Nn, int cin_pts,
                   float* nxy, float* npts, Scratch& S,
                   float* TDF, float* TDH, float* TDH2,
                   int* fpsIdx, float* distWS, hipStream_t s)
{
  const int Bc = 4;
  int M = Nn / 4;
  int ch = cin_pts * 2;
  int cin = cin_pts + 2;

  fps_kernel<<<Bc, 256, 0, s>>>(xy, fpsIdx, distWS, Nn, M);

  size_t txy = (size_t)Bc * M * 2;
  gather_rows_kernel<<<cdivu(txy, 256), 256, 0, s>>>(xy, fpsIdx, nxy, M, 2, Nn, txy);

  knn_kernel<<<cdivu((size_t)Bc * M, 128), 128, 0, s>>>(nxy, xy, S.idx, M, Nn, Bc * M);

  size_t trel = (size_t)Bc * M * 16 * 2;
  gather_relxy_kernel<<<cdivu(trel, 256), 256, 0, s>>>(xy, S.idx, nxy, TDF, M, Nn, cin, +1.f, trel);
  size_t tg = (size_t)Bc * M * 16 * cin_pts;
  gather_feats_kernel<<<cdivu(tg, 256), 256, 0, s>>>(pts, S.idx, TDF, M, Nn, cin_pts, cin, 2, tg);

  int R = Bc * M * 16;
  size_t th = (size_t)R * ch;

  launch_gemm(TDF, cv[0].W, cv[0].b, nullptr, TDH, R, cin, ch, 0, S.Wsw, s);
  bn_stats_kernel<<<ch, 256, 0, s>>>(TDH, R, ch, S.mean, S.rstd);
  bn_apply_relu_kernel<<<cdivu(th, 256), 256, 0, s>>>(TDH, S.mean, S.rstd, cv[0].gamma, cv[0].beta, ch, th);

  launch_gemm(TDH, cv[1].W, cv[1].b, nullptr, TDH2, R, ch, ch, 0, S.Wsw, s);
  bn_stats_kernel<<<ch, 256, 0, s>>>(TDH2, R, ch, S.mean, S.rstd);
  bn_apply_relu_kernel<<<cdivu(th, 256), 256, 0, s>>>(TDH2, S.mean, S.rstd, cv[1].gamma, cv[1].beta, ch, th);

  size_t tm = (size_t)Bc * M * ch;
  maxk_kernel<<<cdivu(tm, 256), 256, 0, s>>>(TDH2, npts, ch, tm);
}

extern "C" void kernel_launch(void* const* d_in, const int* in_sizes, int n_in,
                              void* d_out, int out_size, void* d_ws, size_t ws_size,
                              hipStream_t stream)
{
  (void)in_sizes; (void)n_in; (void)out_size; (void)ws_size;
  const int Bc = 4, N0 = 4096;

  // ---- parameter walk (jax pytree order: sorted dict keys; 'W'<'b'<'beta'<'gamma')
  int ii = 0;
  const float* x = (const float*)d_in[ii++];
  auto nf = [&]() { return (const float*)d_in[ii++]; };
  ConvP conv[4][2];
  TFP btf[4];
  for (int bl = 0; bl < 4; ++bl) {
    for (int cvI = 0; cvI < 2; ++cvI) {
      conv[bl][cvI].W = nf(); conv[bl][cvI].b = nf();
      conv[bl][cvI].beta = nf(); conv[bl][cvI].gamma = nf();
    }
    TFP& t = btf[bl];
    t.delta1.W = nf(); t.delta1.b = nf();
    t.delta2.W = nf(); t.delta2.b = nf();
    t.fc1.W = nf();    t.fc1.b = nf();
    t.fc2.W = nf();    t.fc2.b = nf();
    t.gamma1.W = nf(); t.gamma1.b = nf();
    t.gamma2.W = nf(); t.gamma2.b = nf();
    t.wk = nf(); t.wq = nf(); t.wv = nf();
  }
  LinP fc1a; fc1a.W = nf(); fc1a.b = nf();
  LinP fc1b; fc1b.W = nf(); fc1b.b = nf();
  TFP t1;
  t1.delta1.W = nf(); t1.delta1.b = nf();
  t1.delta2.W = nf(); t1.delta2.b = nf();
  t1.fc1.W = nf();    t1.fc1.b = nf();
  t1.fc2.W = nf();    t1.fc2.b = nf();
  t1.gamma1.W = nf(); t1.gamma1.b = nf();
  t1.gamma2.W = nf(); t1.gamma2.b = nf();
  t1.wk = nf(); t1.wq = nf(); t1.wv = nf();

  // ---- workspace carve
  char* wsb = (char*)d_ws;
  size_t off = 0;
  auto alloc = [&](size_t bytes) -> void* {
    void* p = wsb + off;
    off = (off + bytes + 255) & ~(size_t)255;
    return p;
  };
  const size_t BN0 = (size_t)Bc * N0;           // 16384
  Scratch S;
  S.X   = (float*)alloc(BN0 * 256 * 4);
  S.Q   = (float*)alloc(BN0 * 256 * 4);
  S.KF  = (float*)alloc(BN0 * 256 * 4);
  S.VF  = (float*)alloc(BN0 * 256 * 4);
  S.Rel = (float*)alloc(BN0 * 16 * 2 * 4);
  S.P0  = (float*)alloc(BN0 * 16 * 256 * 4);
  S.P1  = (float*)alloc(BN0 * 16 * 256 * 4);
  S.P2  = (float*)alloc(BN0 * 16 * 256 * 4);
  S.P3  = (float*)alloc(BN0 * 16 * 256 * 4);
  S.idx = (int*)  alloc(BN0 * 16 * 4);
  S.mean = (float*)alloc(512 * 4);
  S.rstd = (float*)alloc(512 * 4);
  S.Wsw  = (bf16_t*)alloc((size_t)512 * 512 * 2 + 4096); // max swizzled weight + pad
  float* xyBuf[2];
  xyBuf[0] = (float*)alloc(BN0 * 2 * 4);
  xyBuf[1] = (float*)alloc(BN0 * 2 * 4);
  float* ptsBuf[3];
  ptsBuf[0] = (float*)alloc(BN0 * 32 * 4);
  ptsBuf[1] = (float*)alloc(BN0 * 32 * 4);
  ptsBuf[2] = (float*)alloc(BN0 * 32 * 4);
  int*   fpsIdx = (int*)  alloc((size_t)Bc * 1024 * 4);
  float* distWS = (float*)alloc(BN0 * 4);
  float* TDF  = (float*)alloc((size_t)Bc * 1024 * 16 * 34 * 4);
  float* TDH  = (float*)alloc((size_t)Bc * 1024 * 16 * 64 * 4);
  float* TDH2 = (float*)alloc((size_t)Bc * 1024 * 16 * 64 * 4);

  // ---- output layout: (pts4, xy0, pts0, xy1, pts1, ..., xy4, pts4)
  const int Ns[5] = {4096, 1024, 256, 64, 16};
  const int Cs[5] = {32, 64, 128, 256, 512};
  float* outF = (float*)d_out;
  size_t head = (size_t)Bc * 16 * 512;   // pts4 at front
  size_t oxy[5], opts[5], cur = head;
  for (int l = 0; l < 5; ++l) {
    oxy[l] = cur;  cur += (size_t)Bc * Ns[l] * 2;
    opts[l] = cur; cur += (size_t)Bc * Ns[l] * Cs[l];
  }

  // ---- level 0
  size_t txy0 = BN0 * 2;
  extract_xy_kernel<<<cdivu(txy0, 256), 256, 0, stream>>>(x, xyBuf[0], 8, txy0);
  launch_gemm(x,   fc1a.W, fc1a.b, nullptr, S.Q,       (int)BN0, 8,  32, 1, S.Wsw, stream); // relu
  launch_gemm(S.Q, fc1b.W, fc1b.b, nullptr, ptsBuf[0], (int)BN0, 32, 32, 0, S.Wsw, stream);
  run_transformer(t1, xyBuf[0], ptsBuf[0], ptsBuf[1], N0, 32, S, stream);
  hipMemcpyAsync(outF + oxy[0],  xyBuf[0], txy0 * 4,      hipMemcpyDeviceToDevice, stream);
  hipMemcpyAsync(outF + opts[0], ptsBuf[1], BN0 * 32 * 4, hipMemcpyDeviceToDevice, stream);

  int curXy = 0, curPts = 1, curN = N0, curC = 32;
  for (int bl = 0; bl < 4; ++bl) {
    int nxyI  = 1 - curXy;
    int tdI   = (curPts + 1) % 3;
    int tfI   = (curPts + 2) % 3;
    run_td(conv[bl], xyBuf[curXy], ptsBuf[curPts], curN, curC,
           xyBuf[nxyI], ptsBuf[tdI], S, TDF, TDH, TDH2, fpsIdx, distWS, stream);
    int M = curN / 4, ch = curC * 2;
    run_transformer(btf[bl], xyBuf[nxyI], ptsBuf[tdI], ptsBuf[tfI], M, ch, S, stream);
    hipMemcpyAsync(outF + oxy[bl + 1],  xyBuf[nxyI], (size_t)Bc * M * 2 * 4,
                   hipMemcpyDeviceToDevice, stream);
    hipMemcpyAsync(outF + opts[bl + 1], ptsBuf[tfI], (size_t)Bc * M * ch * 4,
                   hipMemcpyDeviceToDevice, stream);
    curXy = nxyI; curPts = tfI; curN = M; curC = ch;
  }
  // head copy: pts4 duplicated at offset 0
  hipMemcpyAsync(outF, ptsBuf[curPts], head * 4, hipMemcpyDeviceToDevice, stream);
}
